// GNN_mol_virtual_20641612825181
// MI455X (gfx1250) — compile-verified, matrix operand order checked by reading, not run on hardware
//
#include <hip/hip_runtime.h>
#include <hip/hip_bf16.h>

// ---------------------------------------------------------------------------
// GNN (GCN + virtual node) for MI455X / gfx1250.
//   - Inputs packed once per GEMM into zero-padded bf16 panels (X: [Mpad,320],
//     W: [320,320]) so the GEMM k-loop stages LDS with pure 16B copies
//     (GLOBAL_LOAD_ASYNC_TO_LDS_B128 when available, ASYNCcnt-tracked).
//   - Dense matmuls via v_wmma_f32_16x16x32_bf16, f32 accumulate.
//     One block per 128-row slab computes ALL 19 column tiles -> X read once.
//   - Edge message/scatter via global_atomic_add_f32 (bandwidth bound).
//   - Fused node-update + GroupNorm: 1 block/node, 1 wave32 per group of 30.
// ---------------------------------------------------------------------------

typedef __attribute__((ext_vector_type(16))) __bf16 v16bf;
typedef __attribute__((ext_vector_type(8)))  float  v8f;
typedef __attribute__((vector_size(16)))     int    v4i32;

#define DD 300
#define KPAD 320              // K padded to 10 chunks of 32
#define TILE_K 32
#define ROWS_PER_BLOCK 128    // 8 waves x 16 rows
#define NTILE 19              // ceil(300/16) column tiles
#define ROWB 640              // bf16 panel row stride in bytes (320 * 2)

#if __has_builtin(__builtin_amdgcn_global_load_async_to_lds_b128)
#define HAS_ASYNC_LDS 1
#else
#define HAS_ASYNC_LDS 0
#endif

union ABu { v16bf v; unsigned int u[8]; };

__device__ __forceinline__ unsigned pack_bf16(float x, float y) {
  __bf16 lo = (__bf16)x, hi = (__bf16)y;
  unsigned short ulo, uhi;
  __builtin_memcpy(&ulo, &lo, 2);
  __builtin_memcpy(&uhi, &hi, 2);
  return ((unsigned)uhi << 16) | (unsigned)ulo;
}

__device__ __forceinline__ float warp_sum32(float v) {
  #pragma unroll
  for (int o = 16; o > 0; o >>= 1) v += __shfl_xor(v, o, 32);
  return v;
}

// 16-byte global -> LDS copy; async (ASYNCcnt) on gfx1250.
__device__ __forceinline__ void cp16(void* lds_dst, const void* gsrc) {
#if HAS_ASYNC_LDS
  __builtin_amdgcn_global_load_async_to_lds_b128(
      (__attribute__((address_space(1))) v4i32*)gsrc,
      (__attribute__((address_space(3))) v4i32*)lds_dst,
      /*offset=*/0, /*cpol=*/0);
#else
  *(uint4*)lds_dst = *(const uint4*)gsrc;
#endif
}

__device__ __forceinline__ void cp16_wait() {
#if HAS_ASYNC_LDS
#if __has_builtin(__builtin_amdgcn_s_wait_asynccnt)
  __builtin_amdgcn_s_wait_asynccnt(0);
#else
  asm volatile("s_wait_asynccnt 0" ::: "memory");
#endif
#endif
}

// ---------------- fp32 -> zero-padded bf16 panel packer --------------------
// dst is [rows_pad][160] u32 (= 320 bf16 per row); src is [rows_src][300] f32.
__global__ void k_pack_bf16(const float* __restrict__ src, unsigned* __restrict__ dst,
                            int rows_src, int rows_pad)
{
  int i = blockIdx.x * blockDim.x + threadIdx.x;
  int total = rows_pad * 160;
  if (i >= total) return;
  int r  = i / 160;
  int kp = i - r * 160;
  int k  = kp << 1;
  bool valid = (r < rows_src) && (k < DD);   // pairs never straddle 300 (even)
  size_t idx = valid ? (((size_t)r * DD + k) >> 1) : 0;
  const float2* s2 = (const float2*)src;
  float2 p = s2[idx];
  dst[i] = valid ? pack_bf16(p.x, p.y) : 0u;
}

// ------------------------- WMMA tiled linear -------------------------------
// Y[m, o] = act( (sum_k X[m,k] * W[o,k] + bias[o]) * scale(o) + shift(o) )
// scale/shift implement BatchNorm eval (g * rsqrt(1+eps), b) when bn_g != 0.
// Xbf: [Mpad][320] bf16 (Mpad multiple of 128); Wbf: [320][320] bf16.
// Block = 256 thr (8 waves). Wave w computes rows [row0+16w, row0+16w+16) for
// all 19 col tiles (19 x v8f accumulators). Staging = unguarded 16B copies.
__global__ __launch_bounds__(256)
void k_linear_wmma(const unsigned* __restrict__ Xbf, const unsigned* __restrict__ Wbf,
                   const float* __restrict__ bias,
                   const float* __restrict__ bn_g, const float* __restrict__ bn_b,
                   float* __restrict__ Y, int M, int relu_flag)
{
  __shared__ __align__(16) unsigned aLdsU[ROWS_PER_BLOCK * 16]; // 128 rows x 64B
  __shared__ __align__(16) unsigned bLdsU[320 * 16];            // 320 cols x 64B

  const int row0 = blockIdx.x * ROWS_PER_BLOCK;
  const int tid  = threadIdx.x;
  const int wave = tid >> 5;
  const int lane = tid & 31;
  const int half = lane >> 4;      // K sub-offset +8 for lanes 16..31
  const int l16  = lane & 15;      // M row (A) / N col (B, C/D)

  v8f zero = {};
  v8f acc[NTILE];
  #pragma unroll
  for (int t = 0; t < NTILE; ++t) acc[t] = zero;

  const char* Xb = (const char*)Xbf;
  const char* Wb = (const char*)Wbf;
  char* aB = (char*)aLdsU;
  char* bB = (char*)bLdsU;

  for (int kk = 0; kk < KPAD; kk += TILE_K) {
    const size_t kbyte = (size_t)(kk << 1);       // 64B chunk offset in panel row
    // ---- A chunk: 128 rows x 64B = 8192B; 2 x 16B per thread, no guards ----
    #pragma unroll
    for (int q = 0; q < 2; ++q) {
      int f = tid * 32 + q * 16;
      int r = f >> 6, w = f & 63;
      cp16(aB + f, Xb + (size_t)(row0 + r) * ROWB + kbyte + w);
    }
    // ---- B chunk: 320 cols x 64B = 20480B; 5 x 16B per thread ----
    #pragma unroll
    for (int q = 0; q < 5; ++q) {
      int f = tid * 80 + q * 16;
      int r = f >> 6, w = f & 63;
      cp16(bB + f, Wb + (size_t)r * ROWB + kbyte + w);
    }
    // prefetch next A chunk (global_prefetch_b8)
    if (kk + TILE_K < KPAD)
      __builtin_prefetch(Xb + (size_t)(row0 + (tid >> 1)) * ROWB + kbyte + 64, 0, 3);
    cp16_wait();
    __syncthreads();

    // ---- A fragment for this wave's 16 rows (2 x ds_load_b128) ----
    // ISA 16-bit A 16x32 layout: lanes 0-15 hold K={0..7,16..23},
    // lanes 16-31 hold K={8..15,24..31}; u32 pair index = K/2.
    ABu a;
    const int abase = (((wave << 4) | l16) << 4);
    #pragma unroll
    for (int q = 0; q < 4; ++q) {
      a.u[q]     = aLdsU[abase + (half << 2) + q];
      a.u[4 + q] = aLdsU[abase + 8 + (half << 2) + q];
    }
    // ---- 19 col tiles: B frag (2 x ds_load_b128) + wmma each ----
    #pragma unroll
    for (int t = 0; t < NTILE; ++t) {
      ABu b;
      const int bbase = (((t << 4) | l16) << 4);
      #pragma unroll
      for (int q = 0; q < 4; ++q) {
        b.u[q]     = bLdsU[bbase + (half << 2) + q];
        b.u[4 + q] = bLdsU[bbase + 8 + (half << 2) + q];
      }
      acc[t] = __builtin_amdgcn_wmma_f32_16x16x32_bf16(
          /*neg_a=*/false, a.v, /*neg_b=*/false, b.v,
          /*c_mod=*/(short)0, acc[t], /*reuse_a=*/false, /*reuse_b=*/false);
    }
    __syncthreads();
  }

  // ---- epilogue. C/D: lane<16 -> rows r, lane>=16 -> rows r+8; col = l16 ----
  #pragma unroll
  for (int t = 0; t < NTILE; ++t) {
    int gc = (t << 4) + l16;
    if (gc < DD) {
      float bi = bias ? bias[gc] : 0.f;
      float sc = 1.f, sh = 0.f;
      if (bn_g) { sc = bn_g[gc] * rsqrtf(1.f + 1e-5f); sh = bn_b[gc]; }
      #pragma unroll
      for (int r = 0; r < 8; ++r) {
        int gr = row0 + (wave << 4) + (half ? r + 8 : r);
        if (gr < M) {
          float v = (acc[t][r] + bi) * sc + sh;
          if (relu_flag) v = fmaxf(v, 0.f);
          Y[(size_t)gr * DD + gc] = v;
        }
      }
    }
  }
}

// ------------------------- small utility kernels ---------------------------
__global__ void k_fill(float* __restrict__ p, float v, size_t n) {
  size_t i = (size_t)blockIdx.x * blockDim.x + threadIdx.x;
  size_t stride = (size_t)gridDim.x * blockDim.x;
  for (; i < n; i += stride) p[i] = v;
}

__global__ void k_copy(const float* __restrict__ s, float* __restrict__ d, size_t n) {
  size_t i = (size_t)blockIdx.x * blockDim.x + threadIdx.x;
  size_t stride = (size_t)gridDim.x * blockDim.x;
  for (; i < n; i += stride) d[i] = s[i];
}

__global__ void k_bcast_vn(const float* __restrict__ emb, float* __restrict__ vn, int total) {
  int i = blockIdx.x * blockDim.x + threadIdx.x;
  if (i < total) vn[i] = emb[i % DD];
}

__global__ void k_deg_accum(const int* __restrict__ row, float* __restrict__ deg, int E) {
  int e = blockIdx.x * blockDim.x + threadIdx.x;
  if (e < E) unsafeAtomicAdd(&deg[row[e]], 1.0f);
}

__global__ void k_rsqrt(const float* __restrict__ deg, float* __restrict__ rdeg, int n) {
  int i = blockIdx.x * blockDim.x + threadIdx.x;
  if (i < n) rdeg[i] = rsqrtf(deg[i]);
}

// ------------------------- edge message + scatter --------------------------
// agg[col[e], d] += dinv[row]*dinv[col] * relu(Xp[row[e], d] + ea[e, d])
__global__ __launch_bounds__(320)
void k_edge(const int* __restrict__ row, const int* __restrict__ col,
            const float* __restrict__ ea, const float* __restrict__ Xp,
            const float* __restrict__ rdeg, float* __restrict__ agg, int E)
{
  int e = blockIdx.x;
  int d = threadIdx.x;
  if (e >= E || d >= DD) return;
  int r = row[e], c = col[e];
  float nrm = rdeg[r] * rdeg[c];
  float v = Xp[(size_t)r * DD + d] + ea[(size_t)e * DD + d];
  v = fmaxf(v, 0.f) * nrm;
  unsafeAtomicAdd(&agg[(size_t)c * DD + d], v);
}

// ---------------- fused node update + GroupNorm (+optional ReLU) -----------
// t = agg + relu(Xp + root)/deg + vn[batch];  out = GN(t)*g + b ; relu if flag
// 1 block per node, 10 waves; wave g handles group g (30 features).
__global__ __launch_bounds__(320)
void k_node(const float* __restrict__ Xp, const float* __restrict__ agg,
            const float* __restrict__ deg, const int* __restrict__ batch,
            const float* __restrict__ vn, const float* __restrict__ root,
            const float* __restrict__ gn_g, const float* __restrict__ gn_b,
            float* __restrict__ out, int N, int relu_flag)
{
  int n = blockIdx.x;
  if (n >= N) return;
  int grp  = threadIdx.x >> 5;        // 0..9
  int lane = threadIdx.x & 31;
  int d = grp * 30 + lane;
  bool act = (lane < 30);
  float t = 0.f;
  if (act) {
    float xr = Xp[(size_t)n * DD + d] + root[d];
    t = agg[(size_t)n * DD + d] + fmaxf(xr, 0.f) / deg[n]
        + vn[(size_t)batch[n] * DD + d];
  }
  float s  = warp_sum32(t);
  float s2 = warp_sum32(t * t);
  float mu  = s * (1.f / 30.f);
  float var = s2 * (1.f / 30.f) - mu * mu;
  float hn = (t - mu) * rsqrtf(var + 1e-5f);
  if (act) {
    float v = hn * gn_g[d] + gn_b[d];
    if (relu_flag) v = fmaxf(v, 0.f);
    out[(size_t)n * DD + d] = v;
  }
}

// ---------------- pool layer-input per graph (segment_sum) -----------------
__global__ __launch_bounds__(320)
void k_pool(const float* __restrict__ hin, const int* __restrict__ batch,
            float* __restrict__ pooled, int N)
{
  int n = blockIdx.x;
  int d = threadIdx.x;
  if (n >= N || d >= DD) return;
  unsafeAtomicAdd(&pooled[(size_t)batch[n] * DD + d], hin[(size_t)n * DD + d]);
}

// ---------------------------------------------------------------------------
extern "C" void kernel_launch(void* const* d_in, const int* in_sizes, int n_in,
                              void* d_out, int out_size, void* d_ws, size_t ws_size,
                              hipStream_t stream)
{
  const int N = 100000, E = 200000, D = DD, L = 5, G = 4096;
  const int Npad = ((N + ROWS_PER_BLOCK - 1) / ROWS_PER_BLOCK) * ROWS_PER_BLOCK; // 100096
  const int Gpad = G;                                                             // 4096

  const float* x      = (const float*)d_in[0];
  const int*   ei     = (const int*)d_in[1];
  const float* ea     = (const float*)d_in[2];
  const int*   batch  = (const int*)d_in[3];
  const float* gcn_W  = (const float*)d_in[4];
  const float* gcn_b  = (const float*)d_in[5];
  const float* root   = (const float*)d_in[6];
  const float* gn_g   = (const float*)d_in[7];
  const float* gn_b   = (const float*)d_in[8];
  const float* vn_emb = (const float*)d_in[9];
  const float* vW1    = (const float*)d_in[10];
  const float* vb1    = (const float*)d_in[11];
  const float* vg1    = (const float*)d_in[12];
  const float* vbn1   = (const float*)d_in[13];
  const float* vW2    = (const float*)d_in[14];
  const float* vb2    = (const float*)d_in[15];
  const float* vg2    = (const float*)d_in[16];
  const float* vbn2   = (const float*)d_in[17];

  const int* row = ei;
  const int* col = ei + E;
  float* out = (float*)d_out;

  // ---- workspace carve-out ----
  char* p = (char*)d_ws;
  auto take = [&](size_t bytes) -> void* {
    void* q = (void*)p;
    p += (bytes + 255) & ~(size_t)255;
    return q;
  };
  const size_t NB = (size_t)N * D * sizeof(float);
  const size_t GB = (size_t)G * D * sizeof(float);
  float*    hbuf   = (float*)take(NB);
  float*    Xp     = (float*)take(NB);
  float*    agg    = (float*)take(NB);
  float*    deg    = (float*)take((size_t)N * sizeof(float));
  float*    rdeg   = (float*)take((size_t)N * sizeof(float));
  float*    vnA    = (float*)take(GB);
  float*    vnB    = (float*)take(GB);
  float*    pooled = (float*)take(GB);
  float*    tbuf   = (float*)take(GB);
  unsigned* xbf    = (unsigned*)take((size_t)Npad * 160 * 4);  // bf16 panel (64MB)
  unsigned* wbf    = (unsigned*)take((size_t)320 * 160 * 4);   // bf16 W panel (200KB)

  // ---- degree (deg = 1 + out-degree by row), dinv ----
  k_fill<<<2048, 256, 0, stream>>>(deg, 1.0f, (size_t)N);
  k_deg_accum<<<(E + 255) / 256, 256, 0, stream>>>(row, deg, E);
  k_rsqrt<<<(N + 255) / 256, 256, 0, stream>>>(deg, rdeg, N);

  // ---- virtual node init: broadcast vn_emb to [G, D] ----
  k_bcast_vn<<<(G * D + 255) / 256, 256, 0, stream>>>(vn_emb, vnA, G * D);

  const int packN = (Npad * 160 + 255) / 256;
  const int packG = (Gpad * 160 + 255) / 256;
  const int packW = (320 * 160 + 255) / 256;

  // ping-pong: x -> hbuf -> out -> hbuf -> out -> out (in-place safe at l=4:
  // h_in is only consumed by the pack/GEMM before d_out is overwritten, and
  // there is no pooling after the last layer)
  const float* hin = x;
  float* houts[5] = { hbuf, out, hbuf, out, out };
  float* vncur = vnA;
  float* vnnext = vnB;

  for (int l = 0; l < L; ++l) {
    float* hout = houts[l];

    // Xp = hin @ gcn_W[l]^T + gcn_b[l]   (bf16 panels -> WMMA, f32 accum)
    k_pack_bf16<<<packN, 256, 0, stream>>>(hin, xbf, N, Npad);
    k_pack_bf16<<<packW, 256, 0, stream>>>(gcn_W + (size_t)l * D * D, wbf, DD, 320);
    k_linear_wmma<<<Npad / ROWS_PER_BLOCK, 256, 0, stream>>>(
        xbf, wbf, gcn_b + (size_t)l * D, nullptr, nullptr, Xp, N, 0);

    // agg = scatter_add over edges
    k_fill<<<4096, 256, 0, stream>>>(agg, 0.f, (size_t)N * D);
    k_edge<<<E, 320, 0, stream>>>(row, col, ea, Xp, rdeg, agg, E);

    // h = GN(agg + relu(Xp+root)/deg + vn[batch]) (+ReLU for l < L-1)
    k_node<<<N, 320, 0, stream>>>(Xp, agg, deg, batch, vncur,
                                  root + (size_t)l * D,
                                  gn_g + (size_t)l * D, gn_b + (size_t)l * D,
                                  hout, N, (l < L - 1) ? 1 : 0);

    if (l < L - 1) {
      // pooled = segment_sum(h_in, batch) + vn
      k_copy<<<(G * D + 255) / 256, 256, 0, stream>>>(vncur, pooled, (size_t)G * D);
      k_pool<<<N, 320, 0, stream>>>(hin, batch, pooled, N);

      // vn = relu(bn(relu(bn(pooled@vW1^T+vb1))@vW2^T+vb2))  (WMMA + BN epilogue)
      k_pack_bf16<<<packG, 256, 0, stream>>>(pooled, xbf, G, Gpad);
      k_pack_bf16<<<packW, 256, 0, stream>>>(vW1 + (size_t)l * D * D, wbf, DD, 320);
      k_linear_wmma<<<Gpad / ROWS_PER_BLOCK, 256, 0, stream>>>(
          xbf, wbf, vb1 + (size_t)l * D,
          vg1 + (size_t)l * D, vbn1 + (size_t)l * D, tbuf, G, 1);

      k_pack_bf16<<<packG, 256, 0, stream>>>(tbuf, xbf, G, Gpad);
      k_pack_bf16<<<packW, 256, 0, stream>>>(vW2 + (size_t)l * D * D, wbf, DD, 320);
      k_linear_wmma<<<Gpad / ROWS_PER_BLOCK, 256, 0, stream>>>(
          xbf, wbf, vb2 + (size_t)l * D,
          vg2 + (size_t)l * D, vbn2 + (size_t)l * D, vnnext, G, 1);

      float* tmp = vncur; vncur = vnnext; vnnext = tmp;
    }
    hin = hout;
  }
}
